// RelPositionMultiHeadedAttention_45148696216397
// MI455X (gfx1250) — matrix operand order repaired and checked by reading, not compiled, not run on hardware
//
#include <hip/hip_runtime.h>
#include <hip/hip_bf16.h>

// RelPositionMultiHeadedAttention for MI455X (gfx1250, wave32, WMMA).
// B=4, T=2048, D=256, H=4, dk=64.
// ~35 GFLOP vs ~50MB HBM traffic -> compute bound -> everything on
// v_wmma_f32_16x16x32_f16 (f16 in, f32 accumulate).
//
// Pipeline (all f16 staging in d_ws, ~22MB):
//   proj_kernel x4 : q_u/q_v (B,H,T,dk), k (B,H,T,dk), v^T (B,H,dk,T),
//                    pos (H, 2T+pad, dk)
//   attn_kernel    : flash attention, one wave per 16-row q tile;
//                    rel-shift handled analytically: bd[q,k]=qv[q].p[T-1+k-q]
//   final_kernel   : attn_out @ W_out^T + b_out -> f32 d_out

typedef __attribute__((ext_vector_type(16))) _Float16 v16h;
typedef __attribute__((ext_vector_type(8)))  _Float16 v8h;
typedef __attribute__((ext_vector_type(8)))  float    v8f;
typedef __attribute__((ext_vector_type(4)))  float    v4f;

#define T_LEN  2048
#define D_DIM  256
#define NHEAD  4
#define DK     64
#define BATCH  4

__device__ __forceinline__ v8f vzero8() {
  v8f z;
#pragma unroll
  for (int i = 0; i < 8; ++i) z[i] = 0.0f;
  return z;
}

__device__ __forceinline__ v8f wmma_f16(v16h a, v16h b, v8f c) {
  return __builtin_amdgcn_wmma_f32_16x16x32_f16(
      /*neg_a=*/false, a, /*neg_b=*/false, b,
      /*c_mod=*/(short)0, c, /*reuse_a=*/false, /*reuse_b=*/false);
}

// A fragment (16x32 f16, row-major source). ISA layout: lane<16 -> row=lane,
// K chunks {0..7,16..23}; lane>=16 -> same rows, K chunks {8..15,24..31}.
__device__ __forceinline__ v16h frag_a_f16(const _Float16* __restrict__ base,
                                           int ld, int maxr) {
  const int lane = threadIdx.x & 31;
  int row = lane & 15; if (row > maxr) row = maxr;
  const int k0 = (lane >> 4) << 3;               // 0 or 8
  const _Float16* r = base + row * ld + k0;
  v8h lo = *reinterpret_cast<const v8h*>(r);
  v8h hi = *reinterpret_cast<const v8h*>(r + 16);
  v16h out;
#pragma unroll
  for (int i = 0; i < 8; ++i) { out[i] = lo[i]; out[i + 8] = hi[i]; }
  return out;
}

__device__ __forceinline__ v16h frag_a_f32(const float* __restrict__ base,
                                           int ld, int maxr) {
  const int lane = threadIdx.x & 31;
  int row = lane & 15; if (row > maxr) row = maxr;
  const int k0 = (lane >> 4) << 3;
  const float* r = base + row * ld + k0;
  v4f a0 = *reinterpret_cast<const v4f*>(r);
  v4f a1 = *reinterpret_cast<const v4f*>(r + 4);
  v4f b0 = *reinterpret_cast<const v4f*>(r + 16);
  v4f b1 = *reinterpret_cast<const v4f*>(r + 20);
  v16h out;
#pragma unroll
  for (int i = 0; i < 4; ++i) {
    out[i]      = (_Float16)a0[i];
    out[i + 4]  = (_Float16)a1[i];
    out[i + 8]  = (_Float16)b0[i];
    out[i + 12] = (_Float16)b1[i];
  }
  return out;
}

// B fragment (K=32 x N=16), source is row-major B^T[n][k]. Same K order as A:
// lane<16 -> col=lane, K {0..7,16..23}; lane>=16 -> col=lane-16, K {8..15,24..31}.
__device__ __forceinline__ v16h frag_b_f16(const _Float16* __restrict__ base,
                                           int ld) {
  const int lane = threadIdx.x & 31;
  const int col = lane & 15;
  const int k0 = (lane >> 4) << 3;               // 0 or 8
  const _Float16* r = base + col * ld + k0;
  v8h lo = *reinterpret_cast<const v8h*>(r);
  v8h hi = *reinterpret_cast<const v8h*>(r + 16);
  v16h out;
#pragma unroll
  for (int i = 0; i < 8; ++i) { out[i] = lo[i]; out[i + 8] = hi[i]; }
  return out;
}

__device__ __forceinline__ v16h frag_b_f32(const float* __restrict__ base,
                                           int ld) {
  const int lane = threadIdx.x & 31;
  const int col = lane & 15;
  const int k0 = (lane >> 4) << 3;
  const float* r = base + col * ld + k0;
  v4f a0 = *reinterpret_cast<const v4f*>(r);
  v4f a1 = *reinterpret_cast<const v4f*>(r + 4);
  v4f a2 = *reinterpret_cast<const v4f*>(r + 16);
  v4f a3 = *reinterpret_cast<const v4f*>(r + 20);
  v16h out;
#pragma unroll
  for (int i = 0; i < 4; ++i) {
    out[i]      = (_Float16)a0[i];
    out[i + 4]  = (_Float16)a1[i];
    out[i + 8]  = (_Float16)a2[i];
    out[i + 12] = (_Float16)a3[i];
  }
  return out;
}

// ---------------------------------------------------------------------------
// Projection: Y = X(M,256) @ W(256,256)^T (+bias). One wave = 16 rows x 64
// cols. Block = 4 waves = 16 rows x 256 cols. grid.x = ceil(M/16).
// mode 0: dual out q_u/q_v (B,H,T,dk) with pos_bias_u/v added
// mode 1: out k (B,H,T,dk)
// mode 2: out v^T (B,H,dk,T)
// mode 3: out p (H, 4096, dk), rows >= M zero-padded
// ---------------------------------------------------------------------------
__global__ __launch_bounds__(128) void proj_kernel(
    const float* __restrict__ X, const float* __restrict__ W,
    const float* __restrict__ bias,
    const float* __restrict__ bu, const float* __restrict__ bv,
    _Float16* __restrict__ outU, _Float16* __restrict__ outV,
    int M, int mode) {
  const int lane = threadIdx.x & 31;
  const int rowBase = blockIdx.x * 16;
  if (rowBase >= M) return;                       // wave-uniform
  const int colBase = (threadIdx.x >> 5) * 64;    // wave id -> 64-col tile
  int maxr = M - 1 - rowBase; if (maxr > 15) maxr = 15;

  v8f acc[4];
#pragma unroll
  for (int c = 0; c < 4; ++c) acc[c] = vzero8();

  for (int kk = 0; kk < D_DIM; kk += 32) {
    v16h a = frag_a_f32(X + (size_t)rowBase * D_DIM + kk, D_DIM, maxr);
#pragma unroll
    for (int c = 0; c < 4; ++c) {
      v16h b = frag_b_f32(W + (size_t)(colBase + 16 * c) * D_DIM + kk, D_DIM);
      acc[c] = wmma_f16(a, b, acc[c]);
    }
  }

  const int colIdx = lane & 15;
  const int rh = lane >> 4;
#pragma unroll
  for (int c = 0; c < 4; ++c) {
#pragma unroll
    for (int v = 0; v < 8; ++v) {
      const int row = rowBase + v + 8 * rh;
      const int n = colBase + 16 * c + colIdx;
      float y = acc[c][v];
      if (bias) y += bias[n];
      const int h = n >> 6, d = n & 63;
      if (mode == 3) {                            // pos: (H, 4096, dk)
        const size_t idx = ((size_t)h * 4096 + row) * DK + d;
        outU[idx] = (row < M) ? (_Float16)y : (_Float16)0.0f;
      } else if (row < M) {
        const int b_ = row >> 11, t = row & (T_LEN - 1);
        if (mode == 2) {                          // v^T: (B,H,dk,T)
          const size_t idx = ((size_t)(b_ * NHEAD + h) * DK + d) * T_LEN + t;
          outU[idx] = (_Float16)y;
        } else {                                  // (B,H,T,dk)
          const size_t idx = ((size_t)(b_ * NHEAD + h) * T_LEN + t) * DK + d;
          if (mode == 0) {
            outU[idx] = (_Float16)(y + bu[n]);
            outV[idx] = (_Float16)(y + bv[n]);
          } else {
            outU[idx] = (_Float16)y;
          }
        }
      }
    }
  }
}

// ---------------------------------------------------------------------------
// Flash attention. One wave per (b,h,16-row q tile); 4 waves/block.
// Per 32-col key block: 4 WMMA for ac, 6 WMMA for the rel-pos band
// (bd[q,k] = qv[q] . p[T-1+k-q], diagonal extracted via __shfl),
// online softmax (shfl_xor row reductions), P staged through LDS to become an
// A fragment, 4 WMMA for P@V (V pre-transposed). 14 WMMA / 64 iterations.
// ---------------------------------------------------------------------------
__global__ __launch_bounds__(128) void attn_kernel(
    const _Float16* __restrict__ qu, const _Float16* __restrict__ qv,
    const _Float16* __restrict__ kf, const _Float16* __restrict__ vt,
    const _Float16* __restrict__ pf, const int* __restrict__ mask,
    _Float16* __restrict__ aout) {
  __shared__ __align__(16) _Float16 sP[4][16 * 32];
  const int lane = threadIdx.x & 31;
  const int warp = threadIdx.x >> 5;
  const int wl = blockIdx.x * 4 + warp;           // 0..2047
  const int qTile = wl & 127;
  const int h = (wl >> 7) & (NHEAD - 1);
  const int b = wl >> 9;
  const int q0 = qTile * 16;
  const size_t bh = (size_t)(b * NHEAD + h);

  const _Float16* Qu = qu + (bh * T_LEN + q0) * DK;
  const _Float16* Qv = qv + (bh * T_LEN + q0) * DK;
  const v16h au0 = frag_a_f16(Qu, DK, 15);
  const v16h au1 = frag_a_f16(Qu + 32, DK, 15);
  const v16h av0 = frag_a_f16(Qv, DK, 15);
  const v16h av1 = frag_a_f16(Qv + 32, DK, 15);

  v8f o0 = vzero8(), o1 = vzero8(), o2 = vzero8(), o3 = vzero8();
  float mrow[8], lrow[8];
#pragma unroll
  for (int v = 0; v < 8; ++v) { mrow[v] = -3.0e38f; lrow[v] = 0.0f; }

  const int colIdx = lane & 15;
  const int rh = lane >> 4;
  const _Float16* Kb0 = kf + bh * T_LEN * DK;
  const _Float16* Vb0 = vt + bh * DK * T_LEN;
  const _Float16* Pb0 = pf + (size_t)h * 4096 * DK;
  const int* mptr = mask + b * T_LEN;
  const float scale = 0.125f;                     // 1/sqrt(64)

  for (int kb = 0; kb < T_LEN; kb += 32) {
    const _Float16* Kb = Kb0 + (size_t)kb * DK;
    if (kb + 32 < T_LEN) {                        // global_prefetch_b8
      __builtin_prefetch(Kb + 32 * DK + lane * 64, 0, 1);
      __builtin_prefetch(Vb0 + kb + 32 + (size_t)lane * 2 * T_LEN, 0, 1);
    }
    // --- ac = Q_u . K^T (two 16-col tiles) ---
    v8f s0 = vzero8(), s1 = vzero8();
    s0 = wmma_f16(au0, frag_b_f16(Kb, DK), s0);
    s0 = wmma_f16(au1, frag_b_f16(Kb + 32, DK), s0);
    s1 = wmma_f16(au0, frag_b_f16(Kb + 16 * DK, DK), s1);
    s1 = wmma_f16(au1, frag_b_f16(Kb + 16 * DK + 32, DK), s1);

    // --- rel-pos band: rows j0..j0+47 of p, j0 = T-16+kb-q0 (in bounds) ---
    const _Float16* Pa = Pb0 + (size_t)(T_LEN - 16 + kb - q0) * DK;
    v8f bA = vzero8(), bB = vzero8(), bC = vzero8();
    bA = wmma_f16(av0, frag_b_f16(Pa, DK), bA);
    bA = wmma_f16(av1, frag_b_f16(Pa + 32, DK), bA);
    bB = wmma_f16(av0, frag_b_f16(Pa + 16 * DK, DK), bB);
    bB = wmma_f16(av1, frag_b_f16(Pa + 16 * DK + 32, DK), bB);
    bC = wmma_f16(av0, frag_b_f16(Pa + 32 * DK, DK), bC);
    bC = wmma_f16(av1, frag_b_f16(Pa + 32 * DK + 32, DK), bC);

    const int valid0 = mptr[kb + colIdx];
    const int valid1 = mptr[kb + 16 + colIdx];
    const float vf0 = valid0 ? 1.0f : 0.0f;
    const float vf1 = valid1 ? 1.0f : 0.0f;

#pragma unroll
    for (int v = 0; v < 8; ++v) {
      const int qq = v + 8 * rh;                  // C layout: VGPR v -> row
      const int c = 15 + colIdx - qq;             // band column, 0..30
      const int srcLane = (rh << 4) | (c & 15);   // same row, same half
      const float fA = __shfl(bA[v], srcLane, 32);
      const float fB = __shfl(bB[v], srcLane, 32);
      const float fC = __shfl(bC[v], srcLane, 32);
      const float bd0 = (c < 16) ? fA : fB;
      const float bd1 = (c < 16) ? fB : fC;
      float t0 = (s0[v] + bd0) * scale;
      float t1 = (s1[v] + bd1) * scale;
      t0 = valid0 ? t0 : -1.0e30f;
      t1 = valid1 ? t1 : -1.0e30f;
      // row max across 16 lanes (stays inside each 16-lane half)
      float mx = fmaxf(t0, t1);
      mx = fmaxf(mx, __shfl_xor(mx, 1, 32));
      mx = fmaxf(mx, __shfl_xor(mx, 2, 32));
      mx = fmaxf(mx, __shfl_xor(mx, 4, 32));
      mx = fmaxf(mx, __shfl_xor(mx, 8, 32));
      const float nm = fmaxf(mrow[v], mx);
      const float corr = __expf(mrow[v] - nm);
      mrow[v] = nm;
      const float p0 = __expf(t0 - nm) * vf0;
      const float p1 = __expf(t1 - nm) * vf1;
      float rs = p0 + p1;
      rs += __shfl_xor(rs, 1, 32);
      rs += __shfl_xor(rs, 2, 32);
      rs += __shfl_xor(rs, 4, 32);
      rs += __shfl_xor(rs, 8, 32);
      lrow[v] = lrow[v] * corr + rs;
      o0[v] *= corr; o1[v] *= corr; o2[v] *= corr; o3[v] *= corr;
      sP[warp][qq * 32 + colIdx] = (_Float16)p0;
      sP[warp][qq * 32 + 16 + colIdx] = (_Float16)p1;
    }

    // --- o += P(16x32) @ V(32x64), V stored transposed (dk x T) ---
    const v16h pa = frag_a_f16(&sP[warp][0], 32, 15);
    const _Float16* Vb = Vb0 + kb;
    o0 = wmma_f16(pa, frag_b_f16(Vb, T_LEN), o0);
    o1 = wmma_f16(pa, frag_b_f16(Vb + 16 * T_LEN, T_LEN), o1);
    o2 = wmma_f16(pa, frag_b_f16(Vb + 32 * T_LEN, T_LEN), o2);
    o3 = wmma_f16(pa, frag_b_f16(Vb + 48 * T_LEN, T_LEN), o3);
  }

  // epilogue: normalize and store attn_out[(b,t), h*64+d] as f16
#pragma unroll
  for (int v = 0; v < 8; ++v) {
    const float inv = (lrow[v] > 0.0f) ? (1.0f / lrow[v]) : 0.0f;
    const int qq = v + 8 * rh;
    const size_t base =
        ((size_t)(b * T_LEN + q0 + qq)) * D_DIM + h * DK + colIdx;
    aout[base]      = (_Float16)(o0[v] * inv);
    aout[base + 16] = (_Float16)(o1[v] * inv);
    aout[base + 32] = (_Float16)(o2[v] * inv);
    aout[base + 48] = (_Float16)(o3[v] * inv);
  }
}

// ---------------------------------------------------------------------------
// Final: out(f32) = attn(f16, M x 256) @ W_out^T + b_out
// ---------------------------------------------------------------------------
__global__ __launch_bounds__(128) void final_kernel(
    const _Float16* __restrict__ A, const float* __restrict__ W,
    const float* __restrict__ bias, float* __restrict__ out) {
  const int lane = threadIdx.x & 31;
  const int rowBase = blockIdx.x * 16;
  const int colBase = (threadIdx.x >> 5) * 64;

  v8f acc[4];
#pragma unroll
  for (int c = 0; c < 4; ++c) acc[c] = vzero8();

  for (int kk = 0; kk < D_DIM; kk += 32) {
    v16h a = frag_a_f16(A + (size_t)rowBase * D_DIM + kk, D_DIM, 15);
#pragma unroll
    for (int c = 0; c < 4; ++c) {
      v16h b = frag_b_f32(W + (size_t)(colBase + 16 * c) * D_DIM + kk, D_DIM);
      acc[c] = wmma_f16(a, b, acc[c]);
    }
  }

  const int colIdx = lane & 15;
  const int rh = lane >> 4;
#pragma unroll
  for (int c = 0; c < 4; ++c) {
#pragma unroll
    for (int v = 0; v < 8; ++v) {
      const int row = rowBase + v + 8 * rh;
      const int n = colBase + 16 * c + colIdx;
      out[(size_t)row * D_DIM + n] = acc[c][v] + bias[n];
    }
  }
}

extern "C" void kernel_launch(void* const* d_in, const int* in_sizes, int n_in,
                              void* d_out, int out_size, void* d_ws,
                              size_t ws_size, hipStream_t stream) {
  const float* q_in    = (const float*)d_in[0];
  const float* k_in    = (const float*)d_in[1];
  const float* v_in    = (const float*)d_in[2];
  const float* pos_emb = (const float*)d_in[3];
  const float* W_q     = (const float*)d_in[4];
  const float* b_q     = (const float*)d_in[5];
  const float* W_k     = (const float*)d_in[6];
  const float* b_k     = (const float*)d_in[7];
  const float* W_v     = (const float*)d_in[8];
  const float* b_v     = (const float*)d_in[9];
  const float* W_pos   = (const float*)d_in[10];
  const float* W_out   = (const float*)d_in[11];
  const float* b_out   = (const float*)d_in[12];
  const float* pbu     = (const float*)d_in[13];
  const float* pbv     = (const float*)d_in[14];
  const int*   mask    = (const int*)d_in[15];

  // f16 workspace layout (elements): qu, qv, k, vt: 4*4*2048*64 each;
  // p: 4*4096*64; attn: 4*2048*256. Total ~22MB.
  const size_t SZ_BHTD = (size_t)BATCH * NHEAD * T_LEN * DK;   // 2,097,152
  _Float16* ws = (_Float16*)d_ws;
  _Float16* qu = ws;
  _Float16* qv = qu + SZ_BHTD;
  _Float16* kh = qv + SZ_BHTD;
  _Float16* vt = kh + SZ_BHTD;
  _Float16* ph = vt + SZ_BHTD;
  _Float16* at = ph + (size_t)NHEAD * 4096 * DK;

  const dim3 blk(128);
  const int M = BATCH * T_LEN;                                  // 8192
  proj_kernel<<<M / 16, blk, 0, stream>>>(q_in, W_q, b_q, pbu, pbv,
                                          qu, qv, M, 0);
  proj_kernel<<<M / 16, blk, 0, stream>>>(k_in, W_k, b_k, nullptr, nullptr,
                                          kh, nullptr, M, 1);
  proj_kernel<<<M / 16, blk, 0, stream>>>(v_in, W_v, b_v, nullptr, nullptr,
                                          vt, nullptr, M, 2);
  proj_kernel<<<256, blk, 0, stream>>>(pos_emb, W_pos, nullptr, nullptr,
                                       nullptr, ph, nullptr, 2 * T_LEN - 1, 3);
  attn_kernel<<<(BATCH * NHEAD * (T_LEN / 16)) / 4, blk, 0, stream>>>(
      qu, qv, kh, vt, ph, mask, at);
  final_kernel<<<M / 16, blk, 0, stream>>>(at, W_out, b_out, (float*)d_out);
}
